// SlaterDetSampler_ordered_45243185496083
// MI455X (gfx1250) — 8-bit matrix operand order repaired and checked by reading, compiled, not run on hardware
//
#include <hip/hip_runtime.h>
#include <math.h>

#define DD 96
#define NOCC 48
#define KPAD 64
#define NSLICE 7

typedef int v8i __attribute__((ext_vector_type(8)));

__global__ __launch_bounds__(256) void build_G_kernel(const double* __restrict__ P,
                                                      double* __restrict__ G) {
  __shared__ signed char sl[NSLICE][DD][KPAD];
  const int tid = threadIdx.x;

  for (int idx = tid; idx < DD * KPAD; idx += 256) {
    const int row = idx / KPAD;
    const int k   = idx % KPAD;
    double u = (k < NOCC) ? P[row * NOCC + k] : 0.0;
#pragma unroll
    for (int s = 0; s < NSLICE; ++s) {
      u *= 256.0;
      double c = rint(u);
      if (c > 127.0)  c = 127.0;
      if (c < -127.0) c = -127.0;
      sl[s][row][k] = (signed char)(int)c;
      u -= c;
    }
  }
  __syncthreads();

  const int lane = tid & 31;
  const int wave = tid >> 5;
  const int lg   = lane >> 4;
  const int lm   = lane & 15;

  for (int tile = wave; tile < 36; tile += 8) {
    const int ti = tile / 6;
    const int tj = tile % 6;
    const int arow = ti * 16 + lm;
    const int brow = tj * 16 + lm;

    v8i acc[NSLICE];
#pragma unroll
    for (int p = 0; p < NSLICE; ++p) {
      v8i z = {0, 0, 0, 0, 0, 0, 0, 0};
      acc[p] = z;
    }

#pragma unroll
    for (int s = 0; s < NSLICE; ++s) {
      v8i a;
#pragma unroll
      for (int v = 0; v < 8; ++v) {
        const int kb = ((v >> 1) << 4) + ((v & 1) << 2) + (lg << 3);
        a[v] = *(const int*)&sl[s][arow][kb];
      }
#pragma unroll
      for (int t = 0; t + s < NSLICE; ++t) {
        v8i b;
#pragma unroll
        for (int v = 0; v < 8; ++v) {
          const int kb = ((v >> 1) << 4) + ((v & 1) << 2) + (lg << 3);
          b[v] = *(const int*)&sl[t][brow][kb];
        }
        acc[s + t] = __builtin_amdgcn_wmma_i32_16x16x64_iu8(
             true, a,  true, b, acc[s + t],
             false,  false);
      }
    }

#pragma unroll
    for (int r = 0; r < 8; ++r) {
      double v = 0.0;
#pragma unroll
      for (int p = NSLICE - 1; p >= 0; --p)
        v = v * (1.0 / 256.0) + (double)acc[p][r];
      v *= (1.0 / 65536.0);
      const int i = ti * 16 + r + (lg << 3);
      const int j = tj * 16 + lm;
      G[i * DD + j] = ((i == j) ? 1.0 : 0.0) - v;
    }
  }
}

__global__ __launch_bounds__(256) void minors_kernel(const double* __restrict__ G,
                                                     const int* __restrict__ occ,
                                                     double* __restrict__ out,
                                                     double* __restrict__ pk) {
  const int k   = blockIdx.x;
  const int tid = threadIdx.x;

  __shared__ double T[DD * (DD + 1) / 2];
  __shared__ double piv[DD];
  __shared__ double dfull[DD + 1];
  __shared__ int    socc[NOCC];

  if (tid < NOCC) socc[tid] = occ[tid];

  for (int idx = tid; idx < DD * (DD + 1) / 2; idx += 256) {
    int i = (int)((sqrt(8.0 * (double)idx + 1.0) - 1.0) * 0.5);
    while ((i + 1) * (i + 2) / 2 <= idx) ++i;
    while (i * (i + 1) / 2 > idx) --i;
    const int j = idx - i * (i + 1) / 2;
    T[idx] = G[i * DD + j];
  }
  __syncthreads();

  if (tid < DD) {
    double nb = 0.0;
    for (int m = 0; m < k; ++m)
      if (socc[m] == tid) nb = 1.0;
    T[tid * (tid + 1) / 2 + tid] -= nb;
  }
  __syncthreads();

  for (int j = 0; j < DD; ++j) {
    const double dj = T[j * (j + 1) / 2 + j];
    if (tid == 0) piv[j] = dj;
    const double inv = 1.0 / dj;
    for (int i = j + 1 + tid; i < DD; i += 256)
      T[i * (i + 1) / 2 + j] *= inv;
    __syncthreads();
    for (int i = j + 1; i < DD; ++i) {
      const int c = j + 1 + tid;
      if (c <= i) {
        const int base = i * (i + 1) / 2;
        T[base + c] -= T[base + j] * T[c * (c + 1) / 2 + j] * dj;
      }
    }
    __syncthreads();
  }

  if (tid == 0) {
    dfull[0] = 1.0;
    double p = 1.0;
    for (int j = 0; j < DD; ++j) { p *= piv[j]; dfull[j + 1] = p; }
  }
  __syncthreads();

  if (tid < DD) {
    const int x    = tid;
    const int xmin = (k == 0) ? 0 : (socc[k - 1] + 1);
    const int xmax = DD - NOCC + k + 1;
    const double denom = dfull[xmin];
    const double num   = dfull[x + 1] - dfull[x];
    double p = 0.0;
    if (x >= xmin && x < xmax) p = -num / denom;
    if (fabs(p) <= 1e-15) p = 0.0;
    out[k * DD + x] = p;
    if (x == socc[k]) pk[k] = p;
  }
}

__global__ void prod_kernel(const double* __restrict__ pk, double* __restrict__ out) {
  if (threadIdx.x == 0) {
    double p = 1.0;
    for (int m = 0; m < NOCC; ++m) p *= pk[m];
    out[NOCC * DD] = p;
  }
}

extern "C" void kernel_launch(void* const* d_in, const int* in_sizes, int n_in,
                              void* d_out, int out_size, void* d_ws, size_t ws_size,
                              hipStream_t stream) {
  const double* P   = (const double*)d_in[0];
  const int*    occ = (const int*)d_in[1];
  double* out = (double*)d_out;

  double* G  = (double*)d_ws;
  double* pk = G + DD * DD;

  build_G_kernel<<<1, 256, 0, stream>>>(P, G);
  minors_kernel<<<NOCC, 256, 0, stream>>>(G, occ, out, pk);
  prod_kernel<<<1, 32, 0, stream>>>(pk, out);
}
